// GroupedQueryAttention_33182917329234
// MI455X (gfx1250) — compile-verified, matrix operand order checked by reading, not run on hardware
//
#include <hip/hip_runtime.h>
#include <math.h>

typedef __attribute__((ext_vector_type(16))) __bf16 bf16x16;
typedef __attribute__((ext_vector_type(8)))  __bf16 bf16x8;
typedef __attribute__((ext_vector_type(4)))  __bf16 bf16x4;
typedef __attribute__((ext_vector_type(8)))  float  f32x8;

#define B_  2
#define S_  2048
#define DM_ 2048
#define H_  16
#define KV_ 4
#define HD_ 128

// ---------- CDNA5 async global->LDS copy (guarded: only if toolchain has it) ----------
#if defined(__has_builtin)
#if __has_builtin(__builtin_amdgcn_global_load_async_to_lds_b128)
#define HAVE_ASYNC_LDS 1
#endif
#endif
#ifndef HAVE_ASYNC_LDS
#define HAVE_ASYNC_LDS 0
#endif

#if HAVE_ASYNC_LDS
// Param 1 (per clang diagnostic): 'int __attribute__((vector_size(16))) __device__ *'
//  -> global (AS1) pointer to a 16-byte int vector; param 2 is the LDS (AS3) side.
typedef int v4i_vs __attribute__((vector_size(16)));
typedef __attribute__((address_space(1))) v4i_vs* as1_v4i_t;
typedef __attribute__((address_space(3))) v4i_vs* as3_v4i_t;

static __device__ __forceinline__ void async_copy16(void* lds, const void* g) {
  __builtin_amdgcn_global_load_async_to_lds_b128((as1_v4i_t)g, (as3_v4i_t)lds, 0, 0);
}
static __device__ __forceinline__ void wait_async() {
#if __has_builtin(__builtin_amdgcn_s_wait_asynccnt)
  __builtin_amdgcn_s_wait_asynccnt(0);
#else
  asm volatile("s_wait_asynccnt 0x0" ::: "memory");
#endif
}
#endif

// ---------- helpers ----------
static __device__ __forceinline__ __bf16 f2bf(float f) {
  unsigned u = __builtin_bit_cast(unsigned, f);
  u += 0x7FFFu + ((u >> 16) & 1u);            // round-to-nearest-even
  unsigned short h = (unsigned short)(u >> 16);
  return __builtin_bit_cast(__bf16, h);
}

// Load a 16-bit A/B fragment per ISA 7.12.2: lane holds 8 contiguous K at
// (hl*8) and 8 contiguous K at (16 + hl*8).  Two 16-byte loads.
static __device__ __forceinline__ bf16x16 load_frag(const __bf16* base, int hl) {
  const bf16x8* p = (const bf16x8*)(base + hl * 8);
  bf16x8 lo = p[0];        // K = hl*8 .. hl*8+7
  bf16x8 hi = p[2];        // K = 16 + hl*8 .. +7
  return __builtin_shufflevector(lo, hi, 0,1,2,3,4,5,6,7,8,9,10,11,12,13,14,15);
}

static __device__ __forceinline__ f32x8 wmma_bf16(bf16x16 a, bf16x16 b, f32x8 c) {
  return __builtin_amdgcn_wmma_f32_16x16x32_bf16(false, a, false, b, (short)0, c,
                                                 false, false);
}

// ---------- f32 -> bf16 convert (4 elements / thread, vectorized) ----------
__global__ __launch_bounds__(256) void cvt_bf16_kernel(const float* __restrict__ in,
                                                       __bf16* __restrict__ out, int n) {
  int i = (blockIdx.x * 256 + threadIdx.x) * 4;   // n is always a multiple of 4
  if (i >= n) return;
  float4 v = *(const float4*)&in[i];
  bf16x4 o = {f2bf(v.x), f2bf(v.y), f2bf(v.z), f2bf(v.w)};
  *(bf16x4*)&out[i] = o;
}

// ---------- generic bf16 GEMM, f32 out (row-major A[M,K], B[K,N], C[M,N]) ----------
#define BM 128
#define BN 64
#define BK 64
__global__ __launch_bounds__(256) void gemm_bf16_kernel(const __bf16* __restrict__ A,
                                                        const __bf16* __restrict__ B,
                                                        float* __restrict__ C,
                                                        int M, int N, int K) {
  __shared__ __bf16 As[BM * BK];       // [m][k]   16 KB
  __shared__ __bf16 Bs[BN * BK];       // [n][k]    8 KB (transposed)
  int tid  = threadIdx.x;
  int wid  = tid >> 5;
  int lane = tid & 31;
  int hl   = lane >> 4;
  int l16  = lane & 15;
  int wm   = (wid >> 1) * 32;          // 4 wave rows
  int wn   = (wid & 1) * 32;           // 2 wave cols
  int bm   = blockIdx.y * BM;
  int bn   = blockIdx.x * BN;

  f32x8 acc[2][2];
  acc[0][0] = (f32x8){}; acc[0][1] = (f32x8){};
  acc[1][0] = (f32x8){}; acc[1][1] = (f32x8){};

  const int aR = (tid * 8) >> 6, aC = (tid * 8) & 63;   // A-load coords (per thread)
  const int bK = (tid * 4) >> 6, bN = (tid * 4) & 63;   // B-load coords (per thread)

  for (int k0 = 0; k0 < K; k0 += BK) {
    // ---- stage A tile 128x64 (pure copy: async path if available) ----
#if HAVE_ASYNC_LDS
#pragma unroll
    for (int it = 0; it < 4; ++it) {
      int i = tid * 8 + it * 2048;
      int r = i >> 6, c = i & 63;
      async_copy16(&As[i], &A[(size_t)(bm + r) * K + k0 + c]);
    }
#else
    {
      bf16x8 tA[4];
#pragma unroll
      for (int it = 0; it < 4; ++it) {               // all loads first (MLP)
        int i = tid * 8 + it * 2048;
        int r = i >> 6, c = i & 63;
        tA[it] = *(const bf16x8*)&A[(size_t)(bm + r) * K + k0 + c];
      }
#pragma unroll
      for (int it = 0; it < 4; ++it)
        *(bf16x8*)&As[tid * 8 + it * 2048] = tA[it];
    }
#endif
    // ---- stage B tile 64x64 transposed: batch loads, then scatter stores ----
    {
      bf16x4 tB[4];
#pragma unroll
      for (int it = 0; it < 4; ++it) {
        int i = tid * 4 + it * 1024;
        int kk = i >> 6, n = i & 63;
        tB[it] = *(const bf16x4*)&B[(size_t)(k0 + kk) * N + bn + n];
      }
#pragma unroll
      for (int it = 0; it < 4; ++it) {
        int i = tid * 4 + it * 1024;
        int kk = i >> 6, n = i & 63;
        Bs[(n + 0) * BK + kk] = tB[it][0];
        Bs[(n + 1) * BK + kk] = tB[it][1];
        Bs[(n + 2) * BK + kk] = tB[it][2];
        Bs[(n + 3) * BK + kk] = tB[it][3];
      }
    }
    // prefetch next slab into cache while this one is consumed
    if (k0 + BK < K) {
      __builtin_prefetch(&A[(size_t)(bm + aR) * K + k0 + BK + aC], 0, 0);
      __builtin_prefetch(&B[(size_t)(k0 + BK + bK) * N + bn + bN], 0, 0);
    }
#if HAVE_ASYNC_LDS
    wait_async();
#endif
    __syncthreads();

#pragma unroll
    for (int t = 0; t < 2; ++t) {               // two K=32 steps per slab
      bf16x16 a0 = load_frag(&As[(wm + l16) * BK + t * 32], hl);
      bf16x16 a1 = load_frag(&As[(wm + 16 + l16) * BK + t * 32], hl);
      bf16x16 b0 = load_frag(&Bs[(wn + l16) * BK + t * 32], hl);
      bf16x16 b1 = load_frag(&Bs[(wn + 16 + l16) * BK + t * 32], hl);
      acc[0][0] = wmma_bf16(a0, b0, acc[0][0]);
      acc[0][1] = wmma_bf16(a0, b1, acc[0][1]);
      acc[1][0] = wmma_bf16(a1, b0, acc[1][0]);
      acc[1][1] = wmma_bf16(a1, b1, acc[1][1]);
    }
    __syncthreads();
  }
  // C/D layout: lanes 0-15 rows i, lanes 16-31 rows i+8 (per VGPR i)
#pragma unroll
  for (int mt = 0; mt < 2; ++mt)
#pragma unroll
    for (int nt = 0; nt < 2; ++nt)
#pragma unroll
      for (int i = 0; i < 8; ++i) {
        int row = bm + wm + mt * 16 + hl * 8 + i;
        int col = bn + wn + nt * 16 + l16;
        C[(size_t)row * N + col] = acc[mt][nt][i];
      }
}

// ---------- fused RMSNorm + RoPE, f32 in -> bf16 out [B, NH, S, HD] ----------
__global__ __launch_bounds__(256) void normrope_kernel(const float* __restrict__ in,
                                                       const float* __restrict__ w,
                                                       __bf16* __restrict__ out,
                                                       int NH) {
  int row  = blockIdx.x * 8 + (threadIdx.x >> 5);   // over B*NH*S
  int lane = threadIdx.x & 31;
  int s  = row % S_;
  int bh = row / S_;
  int h  = bh % NH;
  int b  = bh / NH;
  const float* src = in + ((size_t)(b * S_ + s) * NH + h) * HD_;
  float t0 = src[lane], t1 = src[lane + 32], t2 = src[lane + 64], t3 = src[lane + 96];
  float ss = t0 * t0 + t1 * t1 + t2 * t2 + t3 * t3;
#pragma unroll
  for (int m = 16; m >= 1; m >>= 1) ss += __shfl_xor(ss, m, 32);
  float r = rsqrtf(ss * (1.0f / 128.0f) + 1e-6f);
  t0 = t0 * r * w[lane];      t1 = t1 * r * w[lane + 32];
  t2 = t2 * r * w[lane + 64]; t3 = t3 * r * w[lane + 96];
  // RoPE: inv_freq(d) = 10000^(-d/64) for d<64, repeated for d>=64
  float sf = (float)s;
  float a0 = sf * __powf(10000.0f, -((float)lane) / 64.0f);
  float a1 = sf * __powf(10000.0f, -((float)(lane + 32)) / 64.0f);
  float c0 = __cosf(a0), s0 = __sinf(a0);
  float c1 = __cosf(a1), s1 = __sinf(a1);
  float o0 = t0 * c0 - t2 * s0;    // d = lane
  float o1 = t1 * c1 - t3 * s1;    // d = lane+32
  float o2 = t2 * c0 + t0 * s0;    // d = lane+64
  float o3 = t3 * c1 + t1 * s1;    // d = lane+96
  __bf16* dst = out + (size_t)row * HD_;
  dst[lane] = f2bf(o0); dst[lane + 32] = f2bf(o1);
  dst[lane + 64] = f2bf(o2); dst[lane + 96] = f2bf(o3);
}

// ---------- V: [B,S,KV,HD] f32 -> [B,KV,S,HD] bf16 (8 elems / thread) ----------
__global__ __launch_bounds__(256) void vpack_kernel(const float* __restrict__ in,
                                                    __bf16* __restrict__ out) {
  int i = (blockIdx.x * 256 + threadIdx.x) * 8;    // over B*S*KV*HD, d-contiguous
  if (i >= B_ * S_ * KV_ * HD_) return;
  int d = i & (HD_ - 1);
  int g = (i >> 7) & (KV_ - 1);
  int s = (i >> 9) & (S_ - 1);
  int b = i >> 20;
  float4 v0 = *(const float4*)&in[i];
  float4 v1 = *(const float4*)&in[i + 4];
  bf16x8 o = {f2bf(v0.x), f2bf(v0.y), f2bf(v0.z), f2bf(v0.w),
              f2bf(v1.x), f2bf(v1.y), f2bf(v1.z), f2bf(v1.w)};
  *(bf16x8*)&out[(((size_t)(b * KV_ + g) * S_ + s) * HD_) + d] = o;
}

// ---------- flash-style causal GQA attention ----------
// grid: (S/16, KV, B), block: 128 (4 waves = 4 query heads sharing one KV head)
__global__ __launch_bounds__(128) void attn_kernel(const __bf16* __restrict__ Q,   // [B,H,S,HD]
                                                   const __bf16* __restrict__ Kt,  // [B,KV,S,HD]
                                                   const __bf16* __restrict__ Vt,  // [B,KV,S,HD]
                                                   __bf16* __restrict__ O) {       // [B,S,H*HD]
  __shared__ __bf16 Ks[32 * HD_];       // K tile, row-major [key][d]
  __shared__ __bf16 Vts[HD_ * 32];      // V tile transposed [d][key]
  __shared__ __bf16 Ps[4][16 * 32];     // per-wave P scratch [q][key]
  int b = blockIdx.z, g = blockIdx.y, qBase = blockIdx.x * 16;
  int wid  = threadIdx.x >> 5;
  int lane = threadIdx.x & 31;
  int hl   = lane >> 4;
  int l16  = lane & 15;
  int h    = g * 4 + wid;

  // Q fragments: A-layout, loaded straight from global (contiguous in d)
  const __bf16* qp = Q + (((size_t)(b * H_ + h)) * S_ + qBase) * HD_;
  bf16x16 qf[4];
#pragma unroll
  for (int kk = 0; kk < 4; ++kk)
    qf[kk] = load_frag(qp + (size_t)l16 * HD_ + kk * 32, hl);

  f32x8 o[8];
#pragma unroll
  for (int nt = 0; nt < 8; ++nt) o[nt] = (f32x8){};
  float mrow[8], lrow[8];
#pragma unroll
  for (int i = 0; i < 8; ++i) { mrow[i] = -1e30f; lrow[i] = 0.0f; }
  const float scale = 0.08838834764831845f;   // 1/sqrt(128)

  const __bf16* kp = Kt + ((size_t)(b * KV_ + g)) * S_ * HD_;
  const __bf16* vp = Vt + ((size_t)(b * KV_ + g)) * S_ * HD_;
  int nkb = (qBase + 16 + 31) >> 5;           // 32-key blocks (causal bound)

  for (int kb = 0; kb < nkb; ++kb) {
    int keyBase = kb * 32;
    // ---- K tile (pure copy: async path if available) ----
#if HAVE_ASYNC_LDS
#pragma unroll
    for (int it = 0; it < 4; ++it) {
      int i = threadIdx.x * 8 + it * 1024;
      async_copy16(&Ks[i], &kp[(size_t)keyBase * HD_ + i]);
    }
#else
    {
      bf16x8 tK[4];
#pragma unroll
      for (int it = 0; it < 4; ++it)
        tK[it] = *(const bf16x8*)&kp[(size_t)keyBase * HD_ + threadIdx.x * 8 + it * 1024];
#pragma unroll
      for (int it = 0; it < 4; ++it)
        *(bf16x8*)&Ks[threadIdx.x * 8 + it * 1024] = tK[it];
    }
#endif
    // ---- V tile: batch loads, then transpose-scatter into LDS ----
    {
      bf16x8 tV[4];
#pragma unroll
      for (int it = 0; it < 4; ++it) {
        int i = threadIdx.x * 8 + it * 1024;
        int kk = i >> 7, d = i & 127;
        tV[it] = *(const bf16x8*)&vp[((size_t)(keyBase + kk)) * HD_ + d];
      }
#pragma unroll
      for (int it = 0; it < 4; ++it) {
        int i = threadIdx.x * 8 + it * 1024;
        int kk = i >> 7, d = i & 127;
#pragma unroll
        for (int e = 0; e < 8; ++e) Vts[(d + e) * 32 + kk] = tV[it][e];
      }
    }
    // prefetch next K/V tiles
    if (kb + 1 < nkb) {
      __builtin_prefetch(&kp[(size_t)(keyBase + 32) * HD_ + threadIdx.x * 8], 0, 0);
      __builtin_prefetch(&vp[(size_t)(keyBase + 32) * HD_ + threadIdx.x * 8], 0, 0);
    }
#if HAVE_ASYNC_LDS
    wait_async();
#endif
    __syncthreads();

    // scores: two 16x16 tiles over the 32-key block, K-dim = HD = 4 x 32
    f32x8 sc0 = (f32x8){}, sc1 = (f32x8){};
#pragma unroll
    for (int kk = 0; kk < 4; ++kk) {
      bf16x16 kb0 = load_frag(&Ks[(l16) * HD_ + kk * 32], hl);
      bf16x16 kb1 = load_frag(&Ks[(16 + l16) * HD_ + kk * 32], hl);
      sc0 = wmma_bf16(qf[kk], kb0, sc0);
      sc1 = wmma_bf16(qf[kk], kb1, sc1);
    }

    // online softmax per row (row = hl*8 + i in C layout)
#pragma unroll
    for (int i = 0; i < 8; ++i) {
      int qrow = qBase + hl * 8 + i;
      int key0 = keyBase + l16;
      int key1 = keyBase + 16 + l16;
      float x0 = (key0 <= qrow) ? sc0[i] * scale : -1e30f;
      float x1 = (key1 <= qrow) ? sc1[i] * scale : -1e30f;
      float mx = fmaxf(x0, x1);
#pragma unroll
      for (int m = 1; m < 16; m <<= 1) mx = fmaxf(mx, __shfl_xor(mx, m, 32));
      float mn = fmaxf(mrow[i], mx);
      float p0 = __expf(x0 - mn);
      float p1 = __expf(x1 - mn);
      float rs = p0 + p1;
#pragma unroll
      for (int m = 1; m < 16; m <<= 1) rs += __shfl_xor(rs, m, 32);
      float corr = __expf(mrow[i] - mn);
      mrow[i] = mn;
      lrow[i] = lrow[i] * corr + rs;
#pragma unroll
      for (int nt = 0; nt < 8; ++nt) o[nt][i] *= corr;
      // C-layout -> A-layout bounce via per-wave LDS scratch
      Ps[wid][(hl * 8 + i) * 32 + l16]      = f2bf(p0);
      Ps[wid][(hl * 8 + i) * 32 + 16 + l16] = f2bf(p1);
    }

    // P(16x32) x V(32x128): 8 n-tiles
    bf16x16 pf = load_frag(&Ps[wid][l16 * 32], hl);
#pragma unroll
    for (int nt = 0; nt < 8; ++nt) {
      bf16x16 bv = load_frag(&Vts[(nt * 16 + l16) * 32], hl);
      o[nt] = wmma_bf16(pf, bv, o[nt]);
    }
    __syncthreads();
  }

  // normalize + store to [B,S,H*HD] bf16 for the output projection
#pragma unroll
  for (int nt = 0; nt < 8; ++nt)
#pragma unroll
    for (int i = 0; i < 8; ++i) {
      int qrow = qBase + hl * 8 + i;
      float val = o[nt][i] / lrow[i];
      O[((size_t)b * S_ + qrow) * (H_ * HD_) + h * HD_ + nt * 16 + l16] = f2bf(val);
    }
}

// ---------- host launch ----------
extern "C" void kernel_launch(void* const* d_in, const int* in_sizes, int n_in,
                              void* d_out, int out_size, void* d_ws, size_t ws_size,
                              hipStream_t stream) {
  const float* x   = (const float*)d_in[0];
  const float* Wq  = (const float*)d_in[1];
  const float* Wk  = (const float*)d_in[2];
  const float* Wv  = (const float*)d_in[3];
  const float* Wo  = (const float*)d_in[4];
  const float* qnw = (const float*)d_in[5];
  const float* knw = (const float*)d_in[6];

  const int M   = B_ * S_;            // 4096
  const int NQ  = H_ * HD_;           // 2048
  const int NKV = KV_ * HD_;          // 512

  size_t off = 0;
  auto alloc = [&](size_t bytes) {
    void* p = (char*)d_ws + off;
    off += (bytes + 255) & ~(size_t)255;
    return p;
  };
  __bf16* xb   = (__bf16*)alloc((size_t)M * DM_ * 2);
  __bf16* Wqb  = (__bf16*)alloc((size_t)DM_ * NQ * 2);
  __bf16* Wkb  = (__bf16*)alloc((size_t)DM_ * NKV * 2);
  __bf16* Wvb  = (__bf16*)alloc((size_t)DM_ * NKV * 2);
  __bf16* Wob  = (__bf16*)alloc((size_t)NQ * DM_ * 2);
  float*  Qf   = (float*) alloc((size_t)M * NQ * 4);
  float*  Kf   = (float*) alloc((size_t)M * NKV * 4);
  float*  Vf   = (float*) alloc((size_t)M * NKV * 4);
  __bf16* Qb   = (__bf16*)alloc((size_t)M * NQ * 2);
  __bf16* Kb   = (__bf16*)alloc((size_t)M * NKV * 2);
  __bf16* Vb   = (__bf16*)alloc((size_t)M * NKV * 2);
  __bf16* Ob   = (__bf16*)Qf;         // alias: Q f32 is dead after normrope

  // 1) f32 -> bf16 converts (4 elems / thread)
  {
    int n;
    n = M * DM_;    cvt_bf16_kernel<<<(n / 4 + 255) / 256, 256, 0, stream>>>(x,  xb,  n);
    n = DM_ * NQ;   cvt_bf16_kernel<<<(n / 4 + 255) / 256, 256, 0, stream>>>(Wq, Wqb, n);
    n = DM_ * NKV;  cvt_bf16_kernel<<<(n / 4 + 255) / 256, 256, 0, stream>>>(Wk, Wkb, n);
    n = DM_ * NKV;  cvt_bf16_kernel<<<(n / 4 + 255) / 256, 256, 0, stream>>>(Wv, Wvb, n);
    n = NQ * DM_;   cvt_bf16_kernel<<<(n / 4 + 255) / 256, 256, 0, stream>>>(Wo, Wob, n);
  }
  // 2) projections (bf16 WMMA GEMMs, f32 out)
  gemm_bf16_kernel<<<dim3(NQ / BN,  M / BM), 256, 0, stream>>>(xb, Wqb, Qf, M, NQ,  DM_);
  gemm_bf16_kernel<<<dim3(NKV / BN, M / BM), 256, 0, stream>>>(xb, Wkb, Kf, M, NKV, DM_);
  gemm_bf16_kernel<<<dim3(NKV / BN, M / BM), 256, 0, stream>>>(xb, Wvb, Vf, M, NKV, DM_);
  // 3) RMSNorm + RoPE -> bf16 head-major
  normrope_kernel<<<(B_ * H_ * S_) / 8,  256, 0, stream>>>(Qf, qnw, Qb, H_);
  normrope_kernel<<<(B_ * KV_ * S_) / 8, 256, 0, stream>>>(Kf, knw, Kb, KV_);
  // 4) V relayout -> bf16 head-major (8 elems / thread)
  {
    int n = B_ * S_ * KV_ * HD_;
    vpack_kernel<<<(n / 8 + 255) / 256, 256, 0, stream>>>(Vf, Vb);
  }
  // 5) causal GQA attention (WMMA flash)
  attn_kernel<<<dim3(S_ / 16, KV_, B_), 128, 0, stream>>>(Qb, Kb, Vb, Ob);
  // 6) output projection -> f32 d_out
  gemm_bf16_kernel<<<dim3(DM_ / BN, M / BM), 256, 0, stream>>>(Ob, Wob, (float*)d_out,
                                                               M, DM_, NQ);
  (void)in_sizes; (void)n_in; (void)out_size; (void)ws_size;
}
